// TreeResnetConvolution_67619965108944
// MI455X (gfx1250) — matrix operand order repaired and checked by reading, not compiled, not run on hardware
//
#include <hip/hip_runtime.h>
#include <hip/hip_bf16.h>

// ---------------------------------------------------------------------------
// TreeResnetConvolution for MI455X (gfx1250, wave32, WMMA + TDM).
//  - every GEMM on v_wmma_f32_16x16x32_bf16 (fp32 accum)
//  - conv weights pre-converted to bf16 once, then TDM (tensor_load_to_lds)
//    pulls the whole (Cout x 3Cin) weight matrix into LDS per workgroup,
//    overlapped with the per-lane index gather that fills the B tile
//  - B tiles stored fragment-packed in LDS so每 fragment = 2x ds_load_b128
// ---------------------------------------------------------------------------

#define B_N   256
#define T0_N  2048
#define L_N   6144          // gathered length = 3*T0
#define NEG_SLOPE 0.01f
#define EPS_F 1e-5f

typedef __attribute__((ext_vector_type(16))) __bf16          v16bf;
typedef __attribute__((ext_vector_type(8)))  float           v8f;
typedef __attribute__((ext_vector_type(8)))  unsigned short  v8us;
typedef __attribute__((ext_vector_type(4)))  unsigned int    v4u;
typedef __attribute__((ext_vector_type(8)))  int             v8i_t;
typedef __attribute__((ext_vector_type(4)))  int             v4i_t;

__device__ __forceinline__ unsigned short f2bf_bits(float f) {
  unsigned int u = __float_as_uint(f);
  u += 0x7FFFu + ((u >> 16) & 1u);          // round-to-nearest-even
  return (unsigned short)(u >> 16);
}
__device__ __forceinline__ float bf_bits2f(unsigned short u) {
  return __uint_as_float(((unsigned int)u) << 16);
}
__device__ __forceinline__ __bf16 bits2bf(unsigned short u) {
  union { unsigned short s; __bf16 b; } c; c.s = u; return c.b;
}
__device__ __forceinline__ __bf16 f2bf16(float f) { return bits2bf(f2bf_bits(f)); }
__device__ __forceinline__ float leaky(float x) { return x > 0.f ? x : NEG_SLOPE * x; }

__device__ __forceinline__ v8f wmma_bf16(v16bf a, v16bf b, v8f c) {
  return __builtin_amdgcn_wmma_f32_16x16x32_bf16(false, a, false, b, (short)0, c,
                                                 false, false);
}

__device__ __forceinline__ v16bf pack16(v8us lo, v8us hi) {
  union { v8us p[2]; v16bf v; } u;
  u.p[0] = lo; u.p[1] = hi;
  return u.v;
}

// A fragment (16x32) from row-major fp32 (weights in global, small layers only).
__device__ __forceinline__ v16bf fragA_f32(const float* __restrict__ W, int ldk,
                                           int r0, int k0, int lane) {
  int m  = lane & 15;
  int kb = (lane >> 4) << 4;
  const float* row = W + (long long)(r0 + m) * ldk + k0;
  v16bf a;
#pragma unroll
  for (int r = 0; r < 4; ++r) {
    a[2*r]         = f2bf16(row[kb + 2*r]);
    a[2*r + 1]     = f2bf16(row[kb + 2*r + 1]);
    a[8 + 2*r]     = f2bf16(row[8 + kb + 2*r]);
    a[8 + 2*r + 1] = f2bf16(row[8 + kb + 2*r + 1]);
  }
  return a;
}

// A fragment from row-major bf16 weights resident in LDS: two 16B vector loads.
__device__ __forceinline__ v16bf fragA_lds(const unsigned short* __restrict__ s,
                                           int ldk, int r0, int k0, int lane) {
  int m  = lane & 15;
  int kb = (lane >> 4) << 4;                 // 0 or 16
  const unsigned short* row = s + (r0 + m) * ldk + k0 + kb;
  v8us lo = *(const v8us*)(row);
  v8us hi = *(const v8us*)(row + 8);
  return pack16(lo, hi);
}

// B fragment from fragment-packed LDS: 16 contiguous bf16 per lane per chunk.
__device__ __forceinline__ v16bf fragB_packed(const unsigned short* __restrict__ s,
                                              int chunk, int lane) {
  const unsigned short* p = s + (((chunk << 5) + lane) << 4);
  v8us lo = *(const v8us*)(p);
  v8us hi = *(const v8us*)(p + 8);
  return pack16(lo, hi);
}

// B fragment where activations are stored [n][k] row-major bf16 (ld = row len).
__device__ __forceinline__ v16bf fragB_nk(const unsigned short* __restrict__ s,
                                          int ld, int k0, int lane) {
  int n  = lane & 15;
  int kb = (lane >> 4) << 4;
  v16bf b;
#pragma unroll
  for (int r = 0; r < 8; ++r) {
    b[2*r]     = bits2bf(s[n * ld + k0 + kb + 2*r]);
    b[2*r + 1] = bits2bf(s[n * ld + k0 + kb + 2*r + 1]);
  }
  return b;
}

// ---------------------------------------------------------------------------
// Tensor Data Mover: 2-D tile (rows x cols bf16, contiguous) global -> LDS.
// D# per CDNA5 ISA ch.8: group0 = {count/lds_addr/global_addr/type},
// group1 = {data_size, dims, tile dims, strides}. Issued once per workgroup.
// ---------------------------------------------------------------------------
__device__ __forceinline__ void tdm_load_2d_to_lds(const void* gptr,
                                                   unsigned lds_byte_off,
                                                   int rows, int cols) {
  unsigned long long ga = (unsigned long long)gptr;
  v4u g0;
  g0.x = 1u;                                              // count=1, user desc
  g0.y = lds_byte_off;                                    // lds_addr (bytes)
  g0.z = (unsigned)ga;                                    // global_addr[31:0]
  g0.w = (unsigned)((ga >> 32) & 0x01FFFFFFu) | (2u << 30); // [56:32] | type=2
  unsigned d0 = (unsigned)cols, d1 = (unsigned)rows;
  v8i_t g1;
  g1[0] = (int)(1u << 16);                                // data_size=1 -> 2B
  g1[1] = (int)((d0 & 0xFFFFu) << 16);                    // tensor_dim0 lo16
  g1[2] = (int)((d0 >> 16) | ((d1 & 0xFFFFu) << 16));     // dim0 hi | dim1 lo
  g1[3] = (int)((d1 >> 16) | (d0 << 16));                 // dim1 hi | tile_dim0
  g1[4] = (int)(d1 & 0xFFFFu);                            // tile_dim1, tile_dim2=0
  g1[5] = (int)d0;                                        // dim0_stride lo32
  g1[6] = 0;                                              // stride hi, dim1_stride lo
  g1[7] = 0;
  v4i_t z4 = (v4i_t)0;
#if defined(__clang_major__) && (__clang_major__ >= 23)
  v8i_t z8 = (v8i_t)0;
  __builtin_amdgcn_tensor_load_to_lds(g0, g1, z4, z4, z8, 0);
#else
  __builtin_amdgcn_tensor_load_to_lds(g0, g1, z4, z4, 0);
#endif
}

// ---------------------------------------------------------------------------
// Query MLP: (B,512) -> LN/leaky 128 -> LN/leaky 64 -> 32.
// ---------------------------------------------------------------------------
__global__ __launch_bounds__(256) void qmlp_kernel(
    const float* __restrict__ qf,
    const float* __restrict__ w1, const float* __restrict__ b1,
    const float* __restrict__ g1, const float* __restrict__ be1,
    const float* __restrict__ w2, const float* __restrict__ b2,
    const float* __restrict__ g2, const float* __restrict__ be2,
    const float* __restrict__ w3, const float* __restrict__ b3,
    float* __restrict__ q_emb)
{
  __shared__ unsigned short xs[16 * 512];
  __shared__ float          h1[16 * 128];
  __shared__ unsigned short h1b[16 * 128];
  __shared__ float          h2[16 * 64];
  __shared__ unsigned short h2b[16 * 64];
  __shared__ float mu_s[16], iv_s[16];

  const int tid  = threadIdx.x;
  const int wave = tid >> 5, lane = tid & 31;
  const int b0   = blockIdx.x * 16;
  const int n    = lane & 15, half = lane >> 4;

  for (int e = tid; e < 16 * 512; e += 256) {
    int r = e >> 9, c = e & 511;
    xs[e] = f2bf_bits(qf[(b0 + r) * 512 + c]);
  }
  __syncthreads();

  { // L1: 128 features, K=512
    v8f acc = {};
    int f0 = wave * 16;
    for (int k0 = 0; k0 < 512; k0 += 32) {
      acc = wmma_bf16(fragA_f32(w1, 512, f0, k0, lane),
                      fragB_nk(xs, 512, k0, lane), acc);
    }
#pragma unroll
    for (int v = 0; v < 8; ++v) {
      int m = v + half * 8;
      h1[n * 128 + f0 + m] = acc[v] + b1[f0 + m];
    }
  }
  __syncthreads();
  if (tid < 16) {
    float s = 0.f, s2 = 0.f;
    for (int c = 0; c < 128; ++c) { float x = h1[tid * 128 + c]; s += x; s2 += x * x; }
    float mu = s * (1.f / 128.f);
    float var = s2 * (1.f / 128.f) - mu * mu;
    mu_s[tid] = mu; iv_s[tid] = rsqrtf(var + EPS_F);
  }
  __syncthreads();
  for (int e = tid; e < 16 * 128; e += 256) {
    int r = e >> 7, c = e & 127;
    h1b[e] = f2bf_bits(leaky((h1[e] - mu_s[r]) * iv_s[r] * g1[c] + be1[c]));
  }
  __syncthreads();

  if (wave < 4) { // L2: 64 features, K=128
    v8f acc = {};
    int f0 = wave * 16;
    for (int k0 = 0; k0 < 128; k0 += 32) {
      acc = wmma_bf16(fragA_f32(w2, 128, f0, k0, lane),
                      fragB_nk(h1b, 128, k0, lane), acc);
    }
#pragma unroll
    for (int v = 0; v < 8; ++v) {
      int m = v + half * 8;
      h2[n * 64 + f0 + m] = acc[v] + b2[f0 + m];
    }
  }
  __syncthreads();
  if (tid < 16) {
    float s = 0.f, s2 = 0.f;
    for (int c = 0; c < 64; ++c) { float x = h2[tid * 64 + c]; s += x; s2 += x * x; }
    float mu = s * (1.f / 64.f);
    float var = s2 * (1.f / 64.f) - mu * mu;
    mu_s[tid] = mu; iv_s[tid] = rsqrtf(var + EPS_F);
  }
  __syncthreads();
  for (int e = tid; e < 16 * 64; e += 256) {
    int r = e >> 6, c = e & 63;
    h2b[e] = f2bf_bits(leaky((h2[e] - mu_s[r]) * iv_s[r] * g2[c] + be2[c]));
  }
  __syncthreads();

  if (wave < 2) { // L3: 32 features, K=64
    v8f acc = {};
    int f0 = wave * 16;
    for (int k0 = 0; k0 < 64; k0 += 32) {
      acc = wmma_bf16(fragA_f32(w3, 64, f0, k0, lane),
                      fragB_nk(h2b, 64, k0, lane), acc);
    }
#pragma unroll
    for (int v = 0; v < 8; ++v) {
      int m = v + half * 8;
      q_emb[(b0 + n) * 32 + f0 + m] = acc[v] + b3[f0 + m];
    }
  }
}

// ---------------------------------------------------------------------------
__global__ __launch_bounds__(256) void pack_x0_kernel(
    const float* __restrict__ q_emb, const float* __restrict__ trees,
    unsigned short* __restrict__ x0)
{
  long long e = (long long)blockIdx.x * 256 + threadIdx.x;
  const long long total = (long long)B_N * 96 * T0_N;
  if (e >= total) return;
  int l = (int)(e % T0_N);
  int c = (int)((e / T0_N) % 96);
  int b = (int)(e / ((long long)96 * T0_N));
  float v = (c < 32) ? q_emb[b * 32 + c]
                     : trees[((long long)b * 64 + (c - 32)) * T0_N + l];
  x0[e] = f2bf_bits(v);
}

__global__ void zero_kernel(float* __restrict__ p, int nelem) {
  int i = blockIdx.x * blockDim.x + threadIdx.x;
  if (i < nelem) p[i] = 0.f;
}

__global__ void f32_to_bf16_kernel(const float* __restrict__ in,
                                   unsigned short* __restrict__ out, int n) {
  int i = blockIdx.x * 256 + threadIdx.x;
  if (i < n) out[i] = f2bf_bits(in[i]);
}

// ---------------------------------------------------------------------------
// Gather + conv1d(k=3,pad=1) as WMMA GEMM.  Weights (bf16) are TDM-loaded into
// LDS once per workgroup (overlapped with the gather); B tile is written
// fragment-packed so each fragment load is 2x ds_load_b128.
// ---------------------------------------------------------------------------
template <int CIN, int COUT>
__global__ __launch_bounds__(256) void tree_conv_kernel(
    const unsigned short* __restrict__ x, int Lx,
    const long long* __restrict__ idx,
    const unsigned short* __restrict__ Wbf,   // (COUT, 3*CIN) bf16
    const float* __restrict__ bias,
    int stride, int Lout,
    float* __restrict__ y, float* __restrict__ stats)
{
  constexpr int K     = CIN * 3;
  constexpr int NPOS  = (COUT == 64) ? 2 : 1;
  constexpr int KTILE = K * 16;

  __shared__ alignas(32) unsigned short wlds[COUT * K];
  __shared__ alignas(32) unsigned short bs[NPOS * KTILE];
  __shared__ float red[512];

  const int b     = blockIdx.y;
  const int tile0 = blockIdx.x * NPOS;
  const int tid   = threadIdx.x, wave = tid >> 5, lane = tid & 31;

  const long long*      idxb = idx + (long long)b * L_N;
  const unsigned short* xb   = x + (long long)b * CIN * Lx;

  // Kick off TDM weight load (wave 0), then everyone gathers the B tile.
  if (wave == 0) {
    tdm_load_2d_to_lds((const void*)Wbf,
                       (unsigned)(unsigned long long)(const void*)wlds,
                       COUT, K);
  }
  __builtin_prefetch(idxb + (long long)tile0 * 16 * stride, 0, 1);

  for (int e = tid; e < NPOS * KTILE; e += 256) {
    int pt  = e / KTILE;
    int rem = e - pt * KTILE;
    int kk  = rem >> 4, nn = rem & 15;
    int ci  = kk / 3, t = kk - ci * 3;        // W row-major (COUT, CIN, 3)
    int p   = (tile0 + pt) * 16 + nn;
    int g   = p * stride + t - 1;             // pad = 1
    float v = 0.f;
    if (p < Lout && g >= 0 && g < L_N) {
      int src = (int)idxb[g];
      v = bf_bits2f(xb[ci * Lx + src]);
    }
    // fragment-packed: chunk-major, lane-major, 16 contiguous bf16 per lane
    int chunk  = kk >> 5, kr = kk & 31;
    int lane_t = nn | ((kr >> 4) << 4);
    int elem   = kr & 15;
    bs[pt * KTILE + (((chunk << 5) + lane_t) << 4) + elem] = f2bf_bits(v);
  }
  if (wave == 0) __builtin_amdgcn_s_wait_tensorcnt(0);
  __syncthreads();

  const int ct = (NPOS == 2) ? (wave & 3) : wave;   // channel tile
  const int pt = (NPOS == 2) ? (wave >> 2) : 0;     // position tile
  const int c0 = ct * 16;
  const unsigned short* bsp = bs + pt * KTILE;

  v8f acc = {};
#pragma unroll
  for (int c = 0; c < K / 32; ++c) {
    acc = wmma_bf16(fragA_lds(wlds, K, c0, c * 32, lane),
                    fragB_packed(bsp, c, lane), acc);
  }

  const int n = lane & 15, half = lane >> 4;
  const int p = (tile0 + pt) * 16 + n;
  float ls = 0.f, ls2 = 0.f;
  if (p < Lout) {
    float* yb = y + ((long long)b * COUT + c0) * (Lout + 1);
#pragma unroll
    for (int v = 0; v < 8; ++v) {
      int m = v + half * 8;
      float val = acc[v] + bias[c0 + m];
      yb[(long long)m * (Lout + 1) + 1 + p] = val;
      ls += val; ls2 += val * val;
    }
  }
  red[tid] = ls; red[256 + tid] = ls2;
  __syncthreads();
  for (int s = 128; s > 0; s >>= 1) {
    if (tid < s) { red[tid] += red[tid + s]; red[256 + tid] += red[256 + tid + s]; }
    __syncthreads();
  }
  if (tid == 0) {
    atomicAdd(&stats[b * 2 + 0], red[0]);
    atomicAdd(&stats[b * 2 + 1], red[256]);
  }
}

// ---------------------------------------------------------------------------
// standardize (ddof=1, /(std+eps)) + leaky, fp32 -> bf16; l==0 is the
// prepended zero column.
// ---------------------------------------------------------------------------
__global__ __launch_bounds__(256) void finalize_apply_kernel(
    const float* __restrict__ y, const float* __restrict__ stats,
    int Cout, int Lo, unsigned short* __restrict__ xout)
{
  long long e = (long long)blockIdx.x * 256 + threadIdx.x;
  const long long per_b = (long long)Cout * Lo;
  const long long total = (long long)B_N * per_b;
  if (e >= total) return;
  int b = (int)(e / per_b);
  long long r = e - (long long)b * per_b;
  int l = (int)(r % Lo);
  float s = stats[b * 2], s2 = stats[b * 2 + 1];
  float Nf = (float)per_b;
  float mu = s / Nf;
  float var = fmaxf((s2 - Nf * mu * mu) / (Nf - 1.f), 0.f);
  float inv = 1.f / (sqrtf(var) + EPS_F);
  float val = (l == 0) ? 0.f : y[e];
  xout[e] = f2bf_bits(leaky((val - mu) * inv));
}

// maxpool3 commutes with the monotone standardize+leaky.
__global__ __launch_bounds__(256) void maxpool_apply_kernel(
    const float* __restrict__ y, const float* __restrict__ stats,
    unsigned short* __restrict__ xout)
{
  long long e = (long long)blockIdx.x * 256 + threadIdx.x;
  const long long total = (long long)B_N * 64 * T0_N;
  if (e >= total) return;
  int j = (int)(e % T0_N);
  int c = (int)((e / T0_N) % 64);
  int b = (int)(e / ((long long)64 * T0_N));
  const float* yb = y + ((long long)b * 64 + c) * 6145;
  float m = -3.4e38f;
#pragma unroll
  for (int t = 0; t < 3; ++t) {
    int l = 3 * j + t;
    float v = (l == 0) ? 0.f : yb[l];
    m = fmaxf(m, v);
  }
  float s = stats[b * 2], s2 = stats[b * 2 + 1];
  const float Nf = 64.f * 6145.f;
  float mu = s / Nf;
  float var = fmaxf((s2 - Nf * mu * mu) / (Nf - 1.f), 0.f);
  float inv = 1.f / (sqrtf(var) + EPS_F);
  xout[e] = f2bf_bits(leaky((m - mu) * inv));
}

// ---------------------------------------------------------------------------
// Global mean pool + output MLP 128 -> LN/leaky 64 -> LN/leaky 32 -> 1.
// ---------------------------------------------------------------------------
__global__ __launch_bounds__(256) void omlp_kernel(
    const unsigned short* __restrict__ xin, int C, int Lo,
    const float* __restrict__ w1, const float* __restrict__ b1,
    const float* __restrict__ g1, const float* __restrict__ be1,
    const float* __restrict__ w2, const float* __restrict__ b2,
    const float* __restrict__ g2, const float* __restrict__ be2,
    const float* __restrict__ w3, const float* __restrict__ b3,
    float* __restrict__ out)
{
  __shared__ unsigned short pb[16 * 128];
  __shared__ float          h1[16 * 64];
  __shared__ unsigned short h1b[16 * 64];
  __shared__ float          h2[16 * 32];
  __shared__ unsigned short h2b[16 * 32];
  __shared__ float mu_s[16], iv_s[16];

  const int tid = threadIdx.x, wave = tid >> 5, lane = tid & 31;
  const int b0  = blockIdx.x * 16;
  const int n   = lane & 15, half = lane >> 4;

  for (int e = tid; e < 16 * 128; e += 256) {
    int r = e >> 7, c = e & 127;
    const unsigned short* p = xin + ((long long)(b0 + r) * C + c) * Lo;
    float s = 0.f;
    for (int l = 0; l < Lo; ++l) s += bf_bits2f(p[l]);
    pb[e] = f2bf_bits(s / (float)Lo);
  }
  __syncthreads();

  if (wave < 4) { // L1: 64 features, K=128
    v8f acc = {};
    int f0 = wave * 16;
    for (int k0 = 0; k0 < 128; k0 += 32) {
      acc = wmma_bf16(fragA_f32(w1, 128, f0, k0, lane),
                      fragB_nk(pb, 128, k0, lane), acc);
    }
#pragma unroll
    for (int v = 0; v < 8; ++v) { int m = v + half * 8; h1[n * 64 + f0 + m] = acc[v] + b1[f0 + m]; }
  }
  __syncthreads();
  if (tid < 16) {
    float s = 0.f, s2 = 0.f;
    for (int c = 0; c < 64; ++c) { float x = h1[tid * 64 + c]; s += x; s2 += x * x; }
    float mu = s * (1.f / 64.f), var = s2 * (1.f / 64.f) - mu * mu;
    mu_s[tid] = mu; iv_s[tid] = rsqrtf(var + EPS_F);
  }
  __syncthreads();
  for (int e = tid; e < 16 * 64; e += 256) {
    int r = e >> 6, c = e & 63;
    h1b[e] = f2bf_bits(leaky((h1[e] - mu_s[r]) * iv_s[r] * g1[c] + be1[c]));
  }
  __syncthreads();

  if (wave < 2) { // L2: 32 features, K=64
    v8f acc = {};
    int f0 = wave * 16;
    for (int k0 = 0; k0 < 64; k0 += 32) {
      acc = wmma_bf16(fragA_f32(w2, 64, f0, k0, lane),
                      fragB_nk(h1b, 64, k0, lane), acc);
    }
#pragma unroll
    for (int v = 0; v < 8; ++v) { int m = v + half * 8; h2[n * 32 + f0 + m] = acc[v] + b2[f0 + m]; }
  }
  __syncthreads();
  if (tid < 16) {
    float s = 0.f, s2 = 0.f;
    for (int c = 0; c < 32; ++c) { float x = h2[tid * 32 + c]; s += x; s2 += x * x; }
    float mu = s * (1.f / 32.f), var = s2 * (1.f / 32.f) - mu * mu;
    mu_s[tid] = mu; iv_s[tid] = rsqrtf(var + EPS_F);
  }
  __syncthreads();
  for (int e = tid; e < 16 * 32; e += 256) {
    int r = e >> 5, c = e & 31;
    h2b[e] = f2bf_bits(leaky((h2[e] - mu_s[r]) * iv_s[r] * g2[c] + be2[c]));
  }
  __syncthreads();

  if (tid < 16) { // L3: 1 output feature — scalar dot
    float s = b3[0];
    for (int c = 0; c < 32; ++c) s += bf_bits2f(h2b[tid * 32 + c]) * w3[c];
    out[b0 + tid] = s;
  }
}

// ---------------------------------------------------------------------------
// Host launch. params pytree flattened in jax order (sorted keys, depth-first):
//   3..12  omlp: b1,b2,b3,be1,be2,g1,g2,w1,w2,w3
//   13..22 qmlp: b1,b2,b3,be1,be2,g1,g2,w1,w2,w3
//   23..38 res[0..3]: b1,b2,w1,w2 each
//   39,40  tc0: b,w
// ---------------------------------------------------------------------------
extern "C" void kernel_launch(void* const* d_in, const int* in_sizes, int n_in,
                              void* d_out, int out_size, void* d_ws, size_t ws_size,
                              hipStream_t stream)
{
  (void)in_sizes; (void)n_in; (void)out_size; (void)ws_size;

  const float*     qf    = (const float*)d_in[0];
  const float*     trees = (const float*)d_in[1];
  const long long* idx   = (const long long*)d_in[2];

  const float* om_b1 = (const float*)d_in[3];
  const float* om_b2 = (const float*)d_in[4];
  const float* om_b3 = (const float*)d_in[5];
  const float* om_be1= (const float*)d_in[6];
  const float* om_be2= (const float*)d_in[7];
  const float* om_g1 = (const float*)d_in[8];
  const float* om_g2 = (const float*)d_in[9];
  const float* om_w1 = (const float*)d_in[10];
  const float* om_w2 = (const float*)d_in[11];
  const float* om_w3 = (const float*)d_in[12];

  const float* qm_b1 = (const float*)d_in[13];
  const float* qm_b2 = (const float*)d_in[14];
  const float* qm_b3 = (const float*)d_in[15];
  const float* qm_be1= (const float*)d_in[16];
  const float* qm_be2= (const float*)d_in[17];
  const float* qm_g1 = (const float*)d_in[18];
  const float* qm_g2 = (const float*)d_in[19];
  const float* qm_w1 = (const float*)d_in[20];
  const float* qm_w2 = (const float*)d_in[21];
  const float* qm_w3 = (const float*)d_in[22];

  const float* r_b1[4], *r_b2[4], *r_w1[4], *r_w2[4];
  for (int i = 0; i < 4; ++i) {
    r_b1[i] = (const float*)d_in[23 + 4*i + 0];
    r_b2[i] = (const float*)d_in[23 + 4*i + 1];
    r_w1[i] = (const float*)d_in[23 + 4*i + 2];
    r_w2[i] = (const float*)d_in[23 + 4*i + 3];
  }
  const float* tc0_b = (const float*)d_in[39];
  const float* tc0_w = (const float*)d_in[40];

  // Workspace layout
  char* wsb = (char*)d_ws;
  float*          q_emb = (float*)(wsb);
  float*          stats = (float*)(wsb + (64 << 10));
  unsigned short* x0    = (unsigned short*)(wsb + (128 << 10));
  size_t x0_b = (size_t)B_N * 96 * T0_N * 2;
  float*          ybuf  = (float*)(wsb + (128 << 10) + x0_b);
  size_t y_b  = (size_t)B_N * 128 * 6145 * 4;
  unsigned short* xA    = (unsigned short*)((char*)ybuf + y_b);
  size_t x_b  = (size_t)B_N * 128 * 6145 * 2;
  unsigned short* xB    = (unsigned short*)((char*)xA + x_b);
  unsigned short* wpool = (unsigned short*)((char*)xB + x_b);

  // bf16 weight pool offsets (elements)
  unsigned short* wbf_tc0  = wpool;                 // 64*288
  unsigned short* wbf_r0w1 = wbf_tc0  + 64 * 288;   // 64*192
  unsigned short* wbf_r0w2 = wbf_r0w1 + 64 * 192;
  unsigned short* wbf_r1w1 = wbf_r0w2 + 64 * 192;
  unsigned short* wbf_r1w2 = wbf_r1w1 + 64 * 192;
  unsigned short* wbf_r2w1 = wbf_r1w2 + 64 * 192;   // 128*192
  unsigned short* wbf_r2w2 = wbf_r2w1 + 128 * 192;  // 128*384
  unsigned short* wbf_r3w1 = wbf_r2w2 + 128 * 384;
  unsigned short* wbf_r3w2 = wbf_r3w1 + 128 * 384;

  auto cvt = [&](const float* src, unsigned short* dst, int nel) {
    f32_to_bf16_kernel<<<(nel + 255) / 256, 256, 0, stream>>>(src, dst, nel);
  };
  cvt(tc0_w,   wbf_tc0,  64 * 288);
  cvt(r_w1[0], wbf_r0w1, 64 * 192);  cvt(r_w2[0], wbf_r0w2, 64 * 192);
  cvt(r_w1[1], wbf_r1w1, 64 * 192);  cvt(r_w2[1], wbf_r1w2, 64 * 192);
  cvt(r_w1[2], wbf_r2w1, 128 * 192); cvt(r_w2[2], wbf_r2w2, 128 * 384);
  cvt(r_w1[3], wbf_r3w1, 128 * 384); cvt(r_w2[3], wbf_r3w2, 128 * 384);

  // 1) Query MLP, 2) pack x0
  qmlp_kernel<<<B_N / 16, 256, 0, stream>>>(qf,
      qm_w1, qm_b1, qm_g1, qm_be1, qm_w2, qm_b2, qm_g2, qm_be2, qm_w3, qm_b3, q_emb);
  {
    long long tot = (long long)B_N * 96 * T0_N;
    pack_x0_kernel<<<(unsigned)((tot + 255) / 256), 256, 0, stream>>>(q_emb, trees, x0);
  }

  auto pre = [&]() { zero_kernel<<<2, 256, 0, stream>>>(stats, B_N * 2); };
  auto run_apply = [&](int Cout, int Lo, unsigned short* xout) {
    long long tot = (long long)B_N * Cout * Lo;
    finalize_apply_kernel<<<(unsigned)((tot + 255) / 256), 256, 0, stream>>>(
        ybuf, stats, Cout, Lo, xout);
  };

  // 3) tc0 (Cin=96 -> Cout=64) + maxpool-fused standardize/leaky -> xA (64,2048)
  pre();
  tree_conv_kernel<96, 64><<<dim3(192, B_N), 256, 0, stream>>>(
      x0, 2048, idx, wbf_tc0, tc0_b, 1, 6144, ybuf, stats);
  {
    long long tot = (long long)B_N * 64 * T0_N;
    maxpool_apply_kernel<<<(unsigned)((tot + 255) / 256), 256, 0, stream>>>(ybuf, stats, xA);
  }

  // 4) residual blocks (strides 1,1,2,1)
  pre(); tree_conv_kernel<64, 64><<<dim3(192, B_N), 256, 0, stream>>>(
      xA, 2048, idx, wbf_r0w1, r_b1[0], 1, 6144, ybuf, stats); run_apply(64, 6145, xB);
  pre(); tree_conv_kernel<64, 64><<<dim3(192, B_N), 256, 0, stream>>>(
      xB, 6145, idx, wbf_r0w2, r_b2[0], 1, 6144, ybuf, stats); run_apply(64, 6145, xA);

  pre(); tree_conv_kernel<64, 64><<<dim3(192, B_N), 256, 0, stream>>>(
      xA, 6145, idx, wbf_r1w1, r_b1[1], 1, 6144, ybuf, stats); run_apply(64, 6145, xB);
  pre(); tree_conv_kernel<64, 64><<<dim3(192, B_N), 256, 0, stream>>>(
      xB, 6145, idx, wbf_r1w2, r_b2[1], 1, 6144, ybuf, stats); run_apply(64, 6145, xA);

  pre(); tree_conv_kernel<64, 128><<<dim3(192, B_N), 256, 0, stream>>>(
      xA, 6145, idx, wbf_r2w1, r_b1[2], 2, 3072, ybuf, stats); run_apply(128, 3073, xB);
  pre(); tree_conv_kernel<128, 128><<<dim3(384, B_N), 256, 0, stream>>>(
      xB, 3073, idx, wbf_r2w2, r_b2[2], 1, 6144, ybuf, stats); run_apply(128, 6145, xA);

  pre(); tree_conv_kernel<128, 128><<<dim3(384, B_N), 256, 0, stream>>>(
      xA, 6145, idx, wbf_r3w1, r_b1[3], 1, 6144, ybuf, stats); run_apply(128, 6145, xB);
  pre(); tree_conv_kernel<128, 128><<<dim3(384, B_N), 256, 0, stream>>>(
      xB, 6145, idx, wbf_r3w2, r_b2[3], 1, 6144, ybuf, stats); run_apply(128, 6145, xA);

  // 5) mean pool + output MLP
  omlp_kernel<<<B_N / 16, 256, 0, stream>>>(xA, 128, 6145,
      om_w1, om_b1, om_g1, om_be1, om_w2, om_b2, om_g2, om_be2, om_w3, om_b3,
      (float*)d_out);
}
